// SVDHead_27900107555262
// MI455X (gfx1250) — compile-verified, matrix operand order checked by reading, not run on hardware
//
#include <hip/hip_runtime.h>
#include <hip/hip_bf16.h>

typedef __bf16 bf16;
typedef __attribute__((ext_vector_type(16))) __bf16 v16bf;
typedef __attribute__((ext_vector_type(8)))  __bf16 v8bf;
typedef __attribute__((ext_vector_type(8)))  float  v8f;

#define MT   64        // rows per block
#define KC   32        // K chunk (WMMA bf16 K)
#define KP   40        // padded LDS row stride (80B: 16B-aligned, bank-conflict-free)
#define D_IN 1024
#define H0   512
#define H1   256
#define NOUT 9
#define NCH0 (D_IN / KC)   // 32 K-chunks in layer 0
#define XPER ((MT * KC / 2) / 256)   // float2 pairs staged per thread = 4

// ws layout (bf16): W0t [512][1024] | W1t [256][512] | W2t [16][256] (rows 9..15 zero)
#define W0T_ELEMS (H0 * D_IN)
#define W1T_ELEMS (H1 * H0)
#define W2T_ELEMS (16 * H1)
#define PREP_TOTAL (W0T_ELEMS + W1T_ELEMS + W2T_ELEMS)   // 659456 = 2576 * 256

// WMMA bf16 fragment for one lane (ISA 7.12.2): lane l in [0,16) = row/col l,
// K groups {hi*8..} and {16+hi*8..}, hi = lane>=16. Two b128 loads per fragment.
__device__ __forceinline__ v16bf load_frag(const bf16* __restrict__ rowPtr, int hi) {
  v8bf g0 = *(const v8bf*)(rowPtr + hi * 8);
  v8bf g1 = *(const v8bf*)(rowPtr + 16 + hi * 8);
  v16bf r;
#pragma unroll
  for (int i = 0; i < 8; ++i) { r[i] = g0[i]; r[8 + i] = g1[i]; }
  return r;
}

__device__ __forceinline__ v8f wmma_bf16(v16bf a, v16bf b, v8f c) {
  return __builtin_amdgcn_wmma_f32_16x16x32_bf16(false, a, false, b, (short)0, c,
                                                 false, false);
}

// One-time weight convert + transpose into d_ws (runs every launch; deterministic).
__global__ __launch_bounds__(256) void prep_weights(
    const float* __restrict__ W0, const float* __restrict__ W1, const float* __restrict__ W2,
    bf16* __restrict__ W0t, bf16* __restrict__ W1t, bf16* __restrict__ W2t)
{
  int idx = blockIdx.x * 256 + threadIdx.x;
  if (idx < W0T_ELEMS) {                       // W0t[n][k] = W0[k][n]
    int n = idx >> 10, k = idx & (D_IN - 1);
    W0t[idx] = (bf16)W0[(size_t)k * H0 + n];
  } else if (idx < W0T_ELEMS + W1T_ELEMS) {    // W1t[n][k] = W1[k][n]
    int j = idx - W0T_ELEMS;
    int n = j >> 9, k = j & (H0 - 1);
    W1t[j] = (bf16)W1[(size_t)k * H1 + n];
  } else if (idx < PREP_TOTAL) {               // W2t[n][k] = W2[k][n], N padded to 16
    int j = idx - (W0T_ELEMS + W1T_ELEMS);
    int n = j >> 8, k = j & (H1 - 1);
    W2t[j] = (n < NOUT) ? (bf16)W2[(size_t)k * NOUT + n] : (bf16)0.0f;
  }
}

__global__ __launch_bounds__(256) void svdhead_fused(
    const float* __restrict__ x,
    const bf16*  __restrict__ W0t, const float* __restrict__ b0,
    const bf16*  __restrict__ W1t, const float* __restrict__ b1,
    const bf16*  __restrict__ W2t, const float* __restrict__ b2,
    float* __restrict__ out)
{
  __shared__ bf16  h0[MT][H0];        // 64 KB: layer0 -> layer1 activations
  __shared__ bf16  h1[MT][H1];        // 32 KB: layer1 -> layer2 activations
  __shared__ bf16  xb[2][MT][KP];     // 10 KB: double-buffered x chunk
  __shared__ float Rt[MT][12];        // 3 KB: 3x3 heads (padded rows)

  const int tid   = threadIdx.x;
  const int wave  = tid >> 5;
  const int lane  = tid & 31;
  const int l     = lane & 15;
  const int hi    = lane >> 4;
  const int mtile = wave >> 1;        // 4 M-tiles of 16 rows
  const int nhalf = wave & 1;         // 2 waves per M-tile split N
  const int row0  = blockIdx.x * MT;

  // ---- x staging split into (a) global loads -> regs, (b) cvt+pack -> LDS ----
  float2 xr[XPER];
  auto xload = [&](int kc) {
#pragma unroll
    for (int i = 0; i < XPER; ++i) {
      int e = tid + i * 256;
      int r = e >> 4, c2 = (e & 15) * 2;
      xr[i] = *(const float2*)&x[(size_t)(row0 + r) * D_IN + kc * KC + c2];
      if (kc + 2 < NCH0)
        __builtin_prefetch(&x[(size_t)(row0 + r) * D_IN + (kc + 2) * KC + c2], 0, 0);
    }
  };
  auto xstore = [&](int buf) {
#pragma unroll
    for (int i = 0; i < XPER; ++i) {
      int e = tid + i * 256;
      int r = e >> 4, c2 = (e & 15) * 2;
      union { unsigned u; bf16 h[2]; } p;
      p.h[0] = (bf16)xr[i].x; p.h[1] = (bf16)xr[i].y;
      *(unsigned*)&xb[buf][r][c2] = p.u;
    }
  };

  // ---------------- Layer 0: (64x1024)@(1024x512), bias+ReLU ----------------
  v8f acc0[16];
#pragma unroll
  for (int i = 0; i < 16; ++i) acc0[i] = (v8f)0.0f;

  xload(0); xstore(0);
  __syncthreads();
  for (int kc = 0; kc < NCH0; ++kc) {
    const int cur = kc & 1;
    const bool have_next = (kc + 1 < NCH0);
    if (have_next) xload(kc + 1);                 // retire during the WMMA stream

    v16bf a = load_frag(&xb[cur][mtile * 16 + l][0], hi);
    const bf16* bptr = W0t + (size_t)(nhalf * 256 + l) * D_IN + kc * KC;

    // Depth-4 software pipeline: 4 B-fragments in flight over L2 at all times.
    v16bf bq[4];
#pragma unroll
    for (int i = 0; i < 4; ++i) bq[i] = load_frag(bptr + (size_t)i * 16 * D_IN, hi);
#pragma unroll
    for (int nt = 0; nt < 16; ++nt) {
      v16bf bcur = bq[nt & 3];
      if (nt + 4 < 16) bq[nt & 3] = load_frag(bptr + (size_t)(nt + 4) * 16 * D_IN, hi);
      acc0[nt] = wmma_bf16(a, bcur, acc0[nt]);
    }

    if (have_next) xstore(cur ^ 1);               // after loads retired
    __syncthreads();
  }
  // D layout: VGPR v -> (m = v + 8*hi, n = l)
#pragma unroll
  for (int nt = 0; nt < 16; ++nt) {
    int n = nhalf * 256 + nt * 16 + l;
    float bias = b0[n];
#pragma unroll
    for (int v = 0; v < 8; ++v)
      h0[mtile * 16 + v + hi * 8][n] = (bf16)fmaxf(acc0[nt][v] + bias, 0.0f);
  }
  __syncthreads();

  // ------- Layer 1: (64x512)@(512x256), bias+ReLU — barrier-free, pipelined -------
  v8f acc1[8];
#pragma unroll
  for (int i = 0; i < 8; ++i) acc1[i] = (v8f)0.0f;
#pragma unroll 2
  for (int kc = 0; kc < H0 / KC; ++kc) {
    v16bf a = load_frag(&h0[mtile * 16 + l][kc * KC], hi);
    const bf16* bptr = W1t + (size_t)(nhalf * 128 + l) * H0 + kc * KC;
    v16bf bq[4];
#pragma unroll
    for (int i = 0; i < 4; ++i) bq[i] = load_frag(bptr + (size_t)i * 16 * H0, hi);
#pragma unroll
    for (int nt = 0; nt < 8; ++nt) {
      v16bf bcur = bq[nt & 3];
      if (nt + 4 < 8) bq[nt & 3] = load_frag(bptr + (size_t)(nt + 4) * 16 * H0, hi);
      acc1[nt] = wmma_bf16(a, bcur, acc1[nt]);
    }
  }
  __syncthreads();   // h1 WAR vs layer-2 reads below
#pragma unroll
  for (int nt = 0; nt < 8; ++nt) {
    int n = nhalf * 128 + nt * 16 + l;
    float bias = b1[n];
#pragma unroll
    for (int v = 0; v < 8; ++v)
      h1[mtile * 16 + v + hi * 8][n] = (bf16)fmaxf(acc1[nt][v] + bias, 0.0f);
  }
  __syncthreads();

  // ---------------- Layer 2: (64x256)@(256x16-padded), + b2 + eps*I ----------------
  if (nhalf == 0) {                 // one wave per M-tile; uniform branch, EXEC full
    v8f acc2 = (v8f)0.0f;
#pragma unroll
    for (int kc = 0; kc < H1 / KC; ++kc) {
      v16bf a = load_frag(&h1[mtile * 16 + l][kc * KC], hi);
      v16bf b = load_frag(W2t + (size_t)l * H1 + kc * KC, hi);
      acc2 = wmma_bf16(a, b, acc2);
    }
#pragma unroll
    for (int v = 0; v < 8; ++v) {
      if (l < NOUT) {
        float diag = (l == 0 || l == 4 || l == 8) ? 1e-3f : 0.0f;   // + eps*I
        Rt[mtile * 16 + v + hi * 8][l] = acc2[v] + b2[l] + diag;
      }
    }
  }
  __syncthreads();

  // ------- SO(3) projection: U@Vh == orthogonal polar factor (Newton iteration) -------
  if (tid < MT) {
    float q0 = Rt[tid][0], q1 = Rt[tid][1], q2 = Rt[tid][2];
    float q3 = Rt[tid][3], q4 = Rt[tid][4], q5 = Rt[tid][5];
    float q6 = Rt[tid][6], q7 = Rt[tid][7], q8 = Rt[tid][8];
#pragma unroll 1
    for (int it = 0; it < 14; ++it) {       // Q <- 0.5*(Q + cofactor(Q)/det(Q))
      float c0 = q4 * q8 - q5 * q7, c1 = q5 * q6 - q3 * q8, c2 = q3 * q7 - q4 * q6;
      float c3 = q2 * q7 - q1 * q8, c4 = q0 * q8 - q2 * q6, c5 = q1 * q6 - q0 * q7;
      float c6 = q1 * q5 - q2 * q4, c7 = q2 * q3 - q0 * q5, c8 = q0 * q4 - q1 * q3;
      float det = q0 * c0 + q1 * c1 + q2 * c2;
      float id  = 0.5f / det;
      q0 = 0.5f * q0 + c0 * id;  q1 = 0.5f * q1 + c1 * id;  q2 = 0.5f * q2 + c2 * id;
      q3 = 0.5f * q3 + c3 * id;  q4 = 0.5f * q4 + c4 * id;  q5 = 0.5f * q5 + c5 * id;
      q6 = 0.5f * q6 + c6 * id;  q7 = 0.5f * q7 + c7 * id;  q8 = 0.5f * q8 + c8 * id;
    }
    Rt[tid][0] = q0; Rt[tid][1] = q1; Rt[tid][2] = q2;
    Rt[tid][3] = q3; Rt[tid][4] = q4; Rt[tid][5] = q5;
    Rt[tid][6] = q6; Rt[tid][7] = q7; Rt[tid][8] = q8;
  }
  __syncthreads();

  // Fully coalesced block output: 64*9 = 576 contiguous floats.
#pragma unroll
  for (int i = 0; i < 3; ++i) {
    int idx = tid + i * 256;
    if (idx < MT * NOUT)
      out[(size_t)row0 * NOUT + idx] = Rt[idx / NOUT][idx % NOUT];
  }
}

extern "C" void kernel_launch(void* const* d_in, const int* in_sizes, int n_in,
                              void* d_out, int out_size, void* d_ws, size_t ws_size,
                              hipStream_t stream) {
  const float* x  = (const float*)d_in[0];
  const float* W0 = (const float*)d_in[1];
  const float* b0 = (const float*)d_in[2];
  const float* W1 = (const float*)d_in[3];
  const float* b1 = (const float*)d_in[4];
  const float* W2 = (const float*)d_in[5];
  const float* b2 = (const float*)d_in[6];
  float* out = (float*)d_out;

  bf16* W0t = (bf16*)d_ws;
  bf16* W1t = W0t + W0T_ELEMS;
  bf16* W2t = W1t + W1T_ELEMS;

  hipLaunchKernelGGL(prep_weights, dim3(PREP_TOTAL / 256), dim3(256), 0, stream,
                     W0, W1, W2, W0t, W1t, W2t);

  const int rows = in_sizes[0] / D_IN;     // 131072
  hipLaunchKernelGGL(svdhead_fused, dim3(rows / MT), dim3(256), 0, stream,
                     x, W0t, b0, W1t, b1, W2t, b2, out);
  (void)ws_size; (void)n_in; (void)out_size;
}
